// RNNLFR_27049704030509
// MI455X (gfx1250) — compile-verified, matrix operand order checked by reading, not run on hardware
//
#include <hip/hip_runtime.h>

// CDNA5 / gfx1250 — wave32, WMMA 16x16x32 f16
typedef __attribute__((ext_vector_type(16))) _Float16 v16h;
typedef __attribute__((ext_vector_type(8)))  float    v8f;

#define NSTEPS            (1 << 20)
#define CHUNK             256
#define WARM              256
#define NCHUNKS           (NSTEPS / CHUNK)          // 4096
#define CHUNKS_PER_WAVE   16
#define THREADS_PER_BLOCK 256                        // 8 waves
#define CHUNKS_PER_BLOCK  ((THREADS_PER_BLOCK / 32) * CHUNKS_PER_WAVE)  // 128
#define NBLOCKS           (NCHUNKS / CHUNKS_PER_BLOCK)                   // 32

__global__ __launch_bounds__(THREADS_PER_BLOCK)
void rnn_lfr_chunked_wmma(const float* __restrict__ input,
                          const float* __restrict__ A,
                          const float* __restrict__ Bu,
                          const float* __restrict__ Bw,
                          const float* __restrict__ Cy,
                          const float* __restrict__ Dyu,
                          const float* __restrict__ Dyw,
                          const float* __restrict__ Cz,
                          const float* __restrict__ Dzu,
                          const float* __restrict__ fc1_w,
                          const float* __restrict__ fc1_b,
                          const float* __restrict__ fc2_w,
                          const float* __restrict__ fc2_b,
                          float* __restrict__ out)
{
    const int tid   = blockIdx.x * blockDim.x + threadIdx.x;
    const int lane  = threadIdx.x & 31;
    const int wave  = tid >> 5;                 // global wave id
    const int row   = lane & 15;                // column n of the 16x16 tile = chunk slot
    const int chunk = wave * CHUNKS_PER_WAVE + row;
    const int cbase = chunk * CHUNK;            // first output index owned by this chunk
    const int koff  = (lane >= 16) ? 8 : 0;     // K-striping: lanes>=16 hold K 8..15 / 24..31

    // ---- small model parameters (broadcast scalar loads, L2-resident) ----
    const float A00 = A[0],  A01 = A[1],  A10 = A[2],  A11 = A[3];
    const float bu0 = Bu[0], bu1 = Bu[1];
    const float bw0 = Bw[0], bw1 = Bw[1];
    const float Cy0 = Cy[0], Cy1 = Cy[1];
    const float dyu = Dyu[0], DywS = Dyw[0];
    const float Cz0 = Cz[0], Cz1 = Cz[1];
    const float dzu = Dzu[0];
    const float fc2b = fc2_b[0];

    // ---- per-lane slice of the MLP weights, ordered per WMMA f16 operand layout ----
    // K-chunk 0 covers h = 0..31, K-chunk 1 covers h = 32..63.
    // Within a 32-K block, element e of the v16h maps to K = ((e<8)? e : e+8) + koff.
    // Weights quantized to f16 once (WMMA consumes f16 anyway); pre-activation math
    // then runs as packed-f16 VALU (v_pk_fma_f16 / v_pk_max_num_f16).
    v16h w16_0, w16_1, b16_0, b16_1;   // fc1 slices (f16, packed)
    v16h a0, a1;                       // A-operand: fc2_w replicated across all 16 rows
    #pragma unroll
    for (int e = 0; e < 16; ++e) {
        const int k  = ((e < 8) ? e : e + 8) + koff;  // 0..31
        const int h0 = k;          // K-chunk 0
        const int h1 = k + 32;     // K-chunk 1
        w16_0[e] = (_Float16)fc1_w[h0];
        w16_1[e] = (_Float16)fc1_w[h1];
        b16_0[e] = (_Float16)fc1_b[h0];
        b16_1[e] = (_Float16)fc1_b[h1];
        a0[e]    = (_Float16)fc2_w[h0];
        a1[e]    = (_Float16)fc2_w[h1];
    }
    const v16h zero16 = {};
    const v8f  zeroC  = {};

    float x0 = 0.0f, x1 = 0.0f;

    // ================= warm-up phase (no output) =================
    // Uniform trip count across the whole grid => EXEC all-ones for WMMA.
    for (int t = 0; t < WARM; ++t) {
        const int  g     = cbase - WARM + t;
        const bool valid = (g >= 0);              // only false for chunk 0 pre-history
        const float u = valid ? input[g] : 0.0f;
        const float z = Cz0 * x0 + Cz1 * x1 + dzu * u;

        const _Float16 zh = (_Float16)z;
        v16h zs;
        #pragma unroll
        for (int e = 0; e < 16; ++e) zs[e] = zh;

        const v16h bv0 = __builtin_elementwise_max(zs * w16_0 + b16_0, zero16);
        const v16h bv1 = __builtin_elementwise_max(zs * w16_1 + b16_1, zero16);

        // Two INDEPENDENT WMMAs (C = 0 each): no D->C serial hazard between them.
        const v8f acc0 = __builtin_amdgcn_wmma_f32_16x16x32_f16(false, a0, false, bv0,
                                                                (short)0, zeroC, false, false);
        const v8f acc1 = __builtin_amdgcn_wmma_f32_16x16x32_f16(false, a1, false, bv1,
                                                                (short)0, zeroC, false, false);
        const float w = acc0[0] + acc1[0] + fc2b;

        const float nx0 = A00 * x0 + A01 * x1 + bu0 * u + bw0 * w;
        const float nx1 = A10 * x0 + A11 * x1 + bu1 * u + bw1 * w;
        x0 = valid ? nx0 : x0;
        x1 = valid ? nx1 : x1;
    }

    // ================= main phase (emit y every step) =================
    for (int t = 0; t < CHUNK; ++t) {
        const int g = cbase + t;                  // always >= 0
        const float u = input[g];
        const float z = Cz0 * x0 + Cz1 * x1 + dzu * u;

        const _Float16 zh = (_Float16)z;
        v16h zs;
        #pragma unroll
        for (int e = 0; e < 16; ++e) zs[e] = zh;

        const v16h bv0 = __builtin_elementwise_max(zs * w16_0 + b16_0, zero16);
        const v16h bv1 = __builtin_elementwise_max(zs * w16_1 + b16_1, zero16);

        const v8f acc0 = __builtin_amdgcn_wmma_f32_16x16x32_f16(false, a0, false, bv0,
                                                                (short)0, zeroC, false, false);
        const v8f acc1 = __builtin_amdgcn_wmma_f32_16x16x32_f16(false, a1, false, bv1,
                                                                (short)0, zeroC, false, false);
        const float w = acc0[0] + acc1[0] + fc2b;

        // Lanes L and L+16 hold identical state for chunk L: both store the same
        // bit pattern to the same address (deterministic; avoids exec juggling).
        out[g] = Cy0 * x0 + Cy1 * x1 + dyu * u + DywS * w;

        const float nx0 = A00 * x0 + A01 * x1 + bu0 * u + bw0 * w;
        const float nx1 = A10 * x0 + A11 * x1 + bu1 * u + bw1 * w;
        x0 = nx0;
        x1 = nx1;
    }
}

extern "C" void kernel_launch(void* const* d_in, const int* in_sizes, int n_in,
                              void* d_out, int out_size, void* d_ws, size_t ws_size,
                              hipStream_t stream) {
    (void)in_sizes; (void)n_in; (void)out_size; (void)d_ws; (void)ws_size;
    rnn_lfr_chunked_wmma<<<NBLOCKS, THREADS_PER_BLOCK, 0, stream>>>(
        (const float*)d_in[0],   // input
        (const float*)d_in[1],   // A
        (const float*)d_in[2],   // Bu
        (const float*)d_in[3],   // Bw
        (const float*)d_in[4],   // Cy
        (const float*)d_in[5],   // Dyu
        (const float*)d_in[6],   // Dyw
        (const float*)d_in[7],   // Cz
        (const float*)d_in[8],   // Dzu
        (const float*)d_in[9],   // fc1_w
        (const float*)d_in[10],  // fc1_b
        (const float*)d_in[11],  // fc2_w
        (const float*)d_in[12],  // fc2_b
        (float*)d_out);
}
